// SPVCNN_86569360818404
// MI455X (gfx1250) — matrix operand order repaired and checked by reading, not compiled, not run on hardware
//
#include <hip/hip_runtime.h>

typedef __attribute__((ext_vector_type(16))) __bf16 v16bf;
typedef __attribute__((ext_vector_type(8)))  __bf16 v8bf;
typedef __attribute__((ext_vector_type(4)))  __bf16 v4bf;
typedef __attribute__((ext_vector_type(8)))  float  v8f;
typedef __attribute__((ext_vector_type(4)))  float  v4f;
typedef __attribute__((ext_vector_type(4)))  int    v4i;

#define BN_EPS 1e-5f

static __device__ __forceinline__ v8f wmma_bf16(v16bf a, v16bf b, v8f c) {
  return __builtin_amdgcn_wmma_f32_16x16x32_bf16(false, a, false, b, (short)0, c,
                                                 false, false);
}
static __device__ __forceinline__ v16bf cat8(v8bf lo, v8bf hi) {
  return __builtin_shufflevector(lo, hi, 0,1,2,3,4,5,6,7,8,9,10,11,12,13,14,15);
}
static __device__ __forceinline__ v8bf cat4(v4bf lo, v4bf hi) {
  return __builtin_shufflevector(lo, hi, 0,1,2,3,4,5,6,7);
}
static __device__ __forceinline__ v8f zero8f() {
  v8f z;
#pragma unroll
  for (int i = 0; i < 8; ++i) z[i] = 0.f;
  return z;
}

// ---------------------------------------------------------------- utilities
__global__ __launch_bounds__(256) void k_zero(float* __restrict__ p, int n) {
  int i = blockIdx.x * 256 + threadIdx.x;
  if (i < n) p[i] = 0.f;
}
__global__ __launch_bounds__(256) void k_zero_bf(__bf16* __restrict__ p, int n) {
  int i = blockIdx.x * 256 + threadIdx.x;
  if (i < n) p[i] = (__bf16)0.f;
}

__global__ __launch_bounds__(256) void k_vox_sum(const float* __restrict__ pF,
                                                 const int* __restrict__ idxq,
                                                 float* __restrict__ vf, int n4) {
  int i = blockIdx.x * 256 + threadIdx.x;
  if (i < n4) {
    int pt = i >> 2;
    atomicAdd(&vf[(size_t)idxq[pt] * 4 + (i & 3)], pF[i]);
  }
}

__global__ __launch_bounds__(256) void k_vox_div(const float* __restrict__ vf,
                                                 const float* __restrict__ counts,
                                                 __bf16* __restrict__ vfb, int m4) {
  int i = blockIdx.x * 256 + threadIdx.x;
  if (i < m4) vfb[i] = (__bf16)(vf[i] / counts[i >> 2]);
}

__global__ __launch_bounds__(256) void k_bn_relu(const float* __restrict__ x,
                                                 const float* __restrict__ gsum,
                                                 const float* __restrict__ gsq,
                                                 const float* __restrict__ g,
                                                 const float* __restrict__ b,
                                                 __bf16* __restrict__ out, int M) {
  long long i = (long long)blockIdx.x * 256 + threadIdx.x;
  long long nn = (long long)M * 32;
  if (i < nn) {
    int c = (int)(i & 31);
    float inv = 1.f / (float)M;
    float mean = gsum[c] * inv;
    float var = gsq[c] * inv - mean * mean;
    float sc = g[c] * rsqrtf(var + BN_EPS);
    float v = (x[i] - mean) * sc + b[c];
    out[i] = (__bf16)fmaxf(v, 0.f);
  }
}

// ------------------------------------------------- conv1: [M x (27*4)] GEMM
// 27 taps of K=4 packed into 4 zero-padded K=32 WMMA groups.
// Invalid taps redirect to the all-zero row at index M -> unconditional loads.
__global__ __launch_bounds__(256) void k_conv1(const __bf16* __restrict__ vfb,
                                               const int* __restrict__ kmap,
                                               const float* __restrict__ w1,
                                               float* __restrict__ out,
                                               float* __restrict__ gsum,
                                               float* __restrict__ gsq, int M) {
  __shared__ __bf16 BT[4 * 32 * 32];   // [group][co][kk], kk = kl*4+ci, k = 8g+kl
  __shared__ float sS[32], sQ[32];
  const int tid = threadIdx.x;
  for (int e = tid; e < 4096; e += 256) {
    int g = e >> 10, r = e & 1023, co = r >> 5, kk = r & 31;
    int k = g * 8 + (kk >> 2), ci = kk & 3;
    BT[e] = (k < 27) ? (__bf16)w1[(k * 4 + ci) * 32 + co] : (__bf16)0.f;
  }
  if (tid < 32) { sS[tid] = 0.f; sQ[tid] = 0.f; }
  __syncthreads();

  const int lane = tid & 31, wv = tid >> 5;
  const int n = lane & 15, half = lane >> 4;
  const int gw = blockIdx.x * 8 + wv, nw = gridDim.x * 8;
  const int T = (M + 15) >> 4;
  float sA = 0.f, sA2 = 0.f, sB = 0.f, sB2 = 0.f;

  for (int t = gw; t < T; t += nw) {
    const int row0 = t << 4;
    const int row = row0 + n;
    const bool rv = row < M;
    const int* km = kmap + (size_t)(rv ? row : (M - 1)) * 27;

    auto tap = [&](int p) -> int {
      int pc = (p < 27) ? p : 26;
      int id = km[pc];
      return (rv && p < 27 && id >= 0) ? id : M;  // M = zero row
    };
    auto G = [&](int id) -> v4bf { return *(const v4bf*)(vfb + (size_t)id * 4); };

    v8f acc0 = zero8f(), acc1 = zero8f();
#pragma unroll
    for (int g = 0; g < 4; ++g) {
      int p0 = 8 * g + 4 * half;
      v16bf a = cat8(cat4(G(tap(p0)), G(tap(p0 + 1))),
                     cat4(G(tap(p0 + 2)), G(tap(p0 + 3))));
      const __bf16* bp0 = &BT[(g * 32 + n) * 32 + 16 * half];
      const __bf16* bp1 = bp0 + 16 * 32;
      v16bf b0 = cat8(*(const v8bf*)bp0, *(const v8bf*)(bp0 + 8));
      v16bf b1 = cat8(*(const v8bf*)bp1, *(const v8bf*)(bp1 + 8));
      acc0 = wmma_bf16(a, b0, acc0);
      acc1 = wmma_bf16(a, b1, acc1);
    }
#pragma unroll
    for (int r = 0; r < 8; ++r) {
      int rr = row0 + r + 8 * half;
      if (rr < M) {
        float v = acc0[r];
        out[(size_t)rr * 32 + n] = v; sA += v; sA2 += v * v;
        float u = acc1[r];
        out[(size_t)rr * 32 + n + 16] = u; sB += u; sB2 += u * u;
      }
    }
  }
  atomicAdd(&sS[n], sA);  atomicAdd(&sQ[n], sA2);
  atomicAdd(&sS[n + 16], sB); atomicAdd(&sQ[n + 16], sB2);
  __syncthreads();
  if (tid < 32) { atomicAdd(gsum + tid, sS[tid]); atomicAdd(gsq + tid, sQ[tid]); }
}

// -------------------- conv2: 27 taps, K=32 each, 32 voxel rows per wave iter
// Software-pipelined: idx prefetch 2 ahead, A-gather 1 ahead, all loads
// unconditional (invalid -> zero row M of h1). Named scalars only.
__global__ __launch_bounds__(256) void k_conv2(const __bf16* __restrict__ h1,
                                               const int* __restrict__ kmap,
                                               const float* __restrict__ w2,
                                               float* __restrict__ out,
                                               float* __restrict__ gsum,
                                               float* __restrict__ gsq, int M) {
  __shared__ __bf16 BT[27 * 32 * 32];  // [k][co][ci] transposed weights, 54 KB
  __shared__ float sS[32], sQ[32];
  const int tid = threadIdx.x;
  for (int e = tid; e < 27 * 1024; e += 256) {
    int k = e >> 10, r = e & 1023, co = r >> 5, ci = r & 31;
    BT[e] = (__bf16)w2[((size_t)k << 10) + (ci << 5) + co];
  }
  if (tid < 32) { sS[tid] = 0.f; sQ[tid] = 0.f; }
  __syncthreads();

  const int lane = tid & 31, wv = tid >> 5;
  const int n = lane & 15, half = lane >> 4;
  const int gw = blockIdx.x * 8 + wv, nw = gridDim.x * 8;
  const int T2 = (M + 31) >> 5;
  float sA = 0.f, sA2 = 0.f, sB = 0.f, sB2 = 0.f;

  for (int t = gw; t < T2; t += nw) {
    const int row0 = t << 5;
    const int rA = row0 + n, rB = row0 + 16 + n;
    const bool vA = rA < M, vB = rB < M;
    const int* kmA = kmap + (size_t)(vA ? rA : (M - 1)) * 27;
    const int* kmB = kmap + (size_t)(vB ? rB : (M - 1)) * 27;

    auto selA = [&](int k) -> int {
      int id = kmA[k];
      return (vA && id >= 0) ? id : M;  // M = zero row
    };
    auto selB = [&](int k) -> int {
      int id = kmB[k];
      return (vB && id >= 0) ? id : M;
    };
    auto GR = [&](int id) -> v16bf {
      const __bf16* p = h1 + (size_t)id * 32;
      return cat8(*(const v8bf*)(p + 16 * half),
                  *(const v8bf*)(p + 16 * half + 8));
    };

    int nA = selA(1), nB = selB(1);          // idx for k+1
    v16bf aA = GR(selA(0)), aB = GR(selB(0));  // A for k
    v8f a0 = zero8f(), a1 = zero8f(), a2 = zero8f(), a3 = zero8f();

#pragma unroll 3
    for (int k = 0; k < 27; ++k) {
      v16bf cA = aA, cB = aB;
      int gA = nA, gB = nB;
      int k2 = (k + 2 < 27) ? (k + 2) : 26;
      nA = selA(k2);        // prefetch idx for k+2
      nB = selB(k2);
      aA = GR(gA);          // gather A for k+1 (overlaps wmma below)
      aB = GR(gB);
      const __bf16* bp0 = &BT[((unsigned)k * 32 + n) * 32 + 16 * half];
      const __bf16* bp1 = bp0 + 16 * 32;
      v16bf b0 = cat8(*(const v8bf*)bp0, *(const v8bf*)(bp0 + 8));
      v16bf b1 = cat8(*(const v8bf*)bp1, *(const v8bf*)(bp1 + 8));
      a0 = wmma_bf16(cA, b0, a0);
      a1 = wmma_bf16(cA, b1, a1);
      a2 = wmma_bf16(cB, b0, a2);
      a3 = wmma_bf16(cB, b1, a3);
    }
#pragma unroll
    for (int r = 0; r < 8; ++r) {
      int rr = row0 + r + 8 * half;
      if (rr < M) {
        float v = a0[r];
        out[(size_t)rr * 32 + n] = v; sA += v; sA2 += v * v;
        float u = a1[r];
        out[(size_t)rr * 32 + n + 16] = u; sB += u; sB2 += u * u;
      }
    }
#pragma unroll
    for (int r = 0; r < 8; ++r) {
      int rr = row0 + 16 + r + 8 * half;
      if (rr < M) {
        float v = a2[r];
        out[(size_t)rr * 32 + n] = v; sA += v; sA2 += v * v;
        float u = a3[r];
        out[(size_t)rr * 32 + n + 16] = u; sB += u; sB2 += u * u;
      }
    }
  }
  atomicAdd(&sS[n], sA);  atomicAdd(&sQ[n], sA2);
  atomicAdd(&sS[n + 16], sB); atomicAdd(&sQ[n + 16], sB2);
  __syncthreads();
  if (tid < 32) { atomicAdd(gsum + tid, sS[tid]); atomicAdd(gsq + tid, sQ[tid]); }
}

// --------------------------- devoxelize (8-corner) fused with q = p@w_pt+b_pt
// dw is zero wherever a corner is invalid, so gathers are fully branchless.
__global__ __launch_bounds__(256) void k_devox_pt(const __bf16* __restrict__ h2,
                                                  const int* __restrict__ didx,
                                                  const float* __restrict__ dwv,
                                                  const float* __restrict__ w_pt,
                                                  const float* __restrict__ b_pt,
                                                  float* __restrict__ q,
                                                  float* __restrict__ gsum,
                                                  float* __restrict__ gsq, int N) {
  __shared__ __bf16 BT[32 * 32];  // w_pt transposed [co][ci]
  __shared__ float sS[32], sQ[32];
  const int tid = threadIdx.x;
  for (int e = tid; e < 1024; e += 256) {
    int co = e >> 5, ci = e & 31;
    BT[e] = (__bf16)w_pt[ci * 32 + co];
  }
  if (tid < 32) { sS[tid] = 0.f; sQ[tid] = 0.f; }
  __syncthreads();

  const int lane = tid & 31, wv = tid >> 5;
  const int n = lane & 15, half = lane >> 4;
  const int gw = blockIdx.x * 8 + wv, nw = gridDim.x * 8;
  const int T = (N + 15) >> 4;
  const float bp0 = b_pt[n], bp1 = b_pt[n + 16];
  float sA = 0.f, sA2 = 0.f, sB = 0.f, sB2 = 0.f;

  const __bf16* q0 = &BT[n * 32 + 16 * half];
  const __bf16* q1 = q0 + 16 * 32;
  const v16bf b0 = cat8(*(const v8bf*)q0, *(const v8bf*)(q0 + 8));
  const v16bf b1 = cat8(*(const v8bf*)q1, *(const v8bf*)(q1 + 8));

  for (int t = gw; t < T; t += nw) {
    const int row0 = t << 4;
    const int row = row0 + n;
    const int rc = (row < N) ? row : (N - 1);

    v4i d0 = *(const v4i*)(didx + (size_t)rc * 8);
    v4i d1 = *(const v4i*)(didx + (size_t)rc * 8 + 4);
    v4f w0 = *(const v4f*)(dwv + (size_t)rc * 8);
    v4f w1v = *(const v4f*)(dwv + (size_t)rc * 8 + 4);

    float pa[16];
#pragma unroll
    for (int i = 0; i < 16; ++i) pa[i] = 0.f;

    auto corner = [&](int idx, float w) {
      const __bf16* hp = h2 + (size_t)(idx < 0 ? 0 : idx) * 32;
      v8bf lo = *(const v8bf*)(hp + 16 * half);
      v8bf hi = *(const v8bf*)(hp + 16 * half + 8);
#pragma unroll
      for (int j = 0; j < 8; ++j) {
        pa[j] += w * (float)lo[j];
        pa[8 + j] += w * (float)hi[j];
      }
    };
    corner(d0[0], w0[0]); corner(d0[1], w0[1]);
    corner(d0[2], w0[2]); corner(d0[3], w0[3]);
    corner(d1[0], w1v[0]); corner(d1[1], w1v[1]);
    corner(d1[2], w1v[2]); corner(d1[3], w1v[3]);

    v16bf a;
#pragma unroll
    for (int i = 0; i < 16; ++i) a[i] = (__bf16)pa[i];
    v8f acc0, acc1;
#pragma unroll
    for (int r = 0; r < 8; ++r) { acc0[r] = bp0; acc1[r] = bp1; }
    acc0 = wmma_bf16(a, b0, acc0);
    acc1 = wmma_bf16(a, b1, acc1);
#pragma unroll
    for (int r = 0; r < 8; ++r) {
      int rr = row0 + r + 8 * half;
      if (rr < N) {
        float v = acc0[r];
        q[(size_t)rr * 32 + n] = v; sA += v; sA2 += v * v;
        float u = acc1[r];
        q[(size_t)rr * 32 + n + 16] = u; sB += u; sB2 += u * u;
      }
    }
  }
  atomicAdd(&sS[n], sA);  atomicAdd(&sQ[n], sA2);
  atomicAdd(&sS[n + 16], sB); atomicAdd(&sQ[n + 16], sB2);
  __syncthreads();
  if (tid < 32) { atomicAdd(gsum + tid, sS[tid]); atomicAdd(gsq + tid, sQ[tid]); }
}

// ------------------- classifier: relu(bn3(q)) @ w_cls + b_cls  (N x 17 out)
__global__ __launch_bounds__(256) void k_cls(const float* __restrict__ q,
                                             const float* __restrict__ gsum,
                                             const float* __restrict__ gsq,
                                             const float* __restrict__ g3,
                                             const float* __restrict__ b3,
                                             const float* __restrict__ w_cls,
                                             const float* __restrict__ b_cls,
                                             float* __restrict__ out, int N) {
  __shared__ __bf16 BT[32 * 32];  // w_cls transposed [co][ci], co>=17 zero-padded
  const int tid = threadIdx.x;
  for (int e = tid; e < 1024; e += 256) {
    int co = e >> 5, ci = e & 31;
    BT[e] = (co < 17) ? (__bf16)w_cls[ci * 17 + co] : (__bf16)0.f;
  }
  __syncthreads();

  const int lane = tid & 31, wv = tid >> 5;
  const int n = lane & 15, half = lane >> 4;
  const int gw = blockIdx.x * 8 + wv, nw = gridDim.x * 8;
  const int T = (N + 15) >> 4;

  float scl[16], shf[16];
  const float invN = 1.f / (float)N;
#pragma unroll
  for (int j = 0; j < 16; ++j) {
    int c = (j < 8) ? (8 * half + j) : (16 + 8 * half + (j - 8));
    float mean = gsum[c] * invN;
    float var = gsq[c] * invN - mean * mean;
    float s = g3[c] * rsqrtf(var + BN_EPS);
    scl[j] = s;
    shf[j] = b3[c] - mean * s;
  }
  const float bc0 = b_cls[n];
  const float bc1 = (n == 0) ? b_cls[16] : 0.f;

  const __bf16* p0 = &BT[n * 32 + 8 * half];
  const __bf16* p1 = p0 + 16 * 32;
  const v16bf b0 = cat8(*(const v8bf*)p0, *(const v8bf*)(p0 + 16));
  const v16bf b1 = cat8(*(const v8bf*)p1, *(const v8bf*)(p1 + 16));

  for (int t = gw; t < T; t += nw) {
    const int row0 = t << 4;
    const int row = row0 + n;
    const int rc = (row < N) ? row : (N - 1);
    const float* qp = q + (size_t)rc * 32;
    v4f x0 = *(const v4f*)(qp + 8 * half);
    v4f x1 = *(const v4f*)(qp + 8 * half + 4);
    v4f x2 = *(const v4f*)(qp + 16 + 8 * half);
    v4f x3 = *(const v4f*)(qp + 16 + 8 * half + 4);
    float xs[16];
#pragma unroll
    for (int j = 0; j < 4; ++j) {
      xs[j] = x0[j]; xs[4 + j] = x1[j]; xs[8 + j] = x2[j]; xs[12 + j] = x3[j];
    }
    v16bf a;
#pragma unroll
    for (int j = 0; j < 16; ++j)
      a[j] = (__bf16)fmaxf(xs[j] * scl[j] + shf[j], 0.f);

    v8f acc0, acc1;
#pragma unroll
    for (int r = 0; r < 8; ++r) { acc0[r] = bc0; acc1[r] = bc1; }
    acc0 = wmma_bf16(a, b0, acc0);
    acc1 = wmma_bf16(a, b1, acc1);
#pragma unroll
    for (int r = 0; r < 8; ++r) {
      int rr = row0 + r + 8 * half;
      if (rr < N) {
        out[(size_t)rr * 17 + n] = acc0[r];
        if (n == 0) out[(size_t)rr * 17 + 16] = acc1[r];
      }
    }
  }
}

// ---------------------------------------------------------------- launcher
extern "C" void kernel_launch(void* const* d_in, const int* in_sizes, int n_in,
                              void* d_out, int out_size, void* d_ws, size_t ws_size,
                              hipStream_t stream) {
  const float* pF     = (const float*)d_in[0];
  const float* counts = (const float*)d_in[1];
  const float* dwv    = (const float*)d_in[2];
  const float* w1     = (const float*)d_in[3];
  const float* g1     = (const float*)d_in[4];
  const float* b1     = (const float*)d_in[5];
  const float* w2     = (const float*)d_in[6];
  const float* g2     = (const float*)d_in[7];
  const float* b2     = (const float*)d_in[8];
  const float* w_pt   = (const float*)d_in[9];
  const float* b_pt   = (const float*)d_in[10];
  const float* g3     = (const float*)d_in[11];
  const float* b3     = (const float*)d_in[12];
  const float* w_cls  = (const float*)d_in[13];
  const float* b_cls  = (const float*)d_in[14];
  const int* idxq     = (const int*)d_in[15];
  const int* kmap     = (const int*)d_in[16];
  const int* didx     = (const int*)d_in[17];

  const int N = in_sizes[0] / 4;  // points
  const int M = in_sizes[1];      // voxels
  float* out = (float*)d_out;

  char* ws = (char*)d_ws;
  auto rnd = [](size_t x) { return (x + 255) & ~(size_t)255; };
  size_t o = 0;
  float* stats = (float*)(ws + o); o += 1024;                        // 6x32 fp32
  float* vf    = (float*)(ws + o); o += rnd((size_t)M * 4 * 4);      // fp32 accum
  __bf16* vfb  = (__bf16*)(ws + o); o += rnd((size_t)(M + 1) * 4 * 2);   // + zero row
  float* big   = (float*)(ws + o); o += rnd((size_t)N * 32 * 4);     // c1 -> c2 -> q
  __bf16* h1   = (__bf16*)(ws + o); o += rnd((size_t)(M + 1) * 32 * 2);  // + zero row
  __bf16* h2   = (__bf16*)(ws + o); o += rnd((size_t)M * 32 * 2);

  k_zero<<<1, 256, 0, stream>>>(stats, 256);
  k_zero<<<(M * 4 + 255) / 256, 256, 0, stream>>>(vf, M * 4);
  k_zero_bf<<<1, 256, 0, stream>>>(vfb + (size_t)M * 4, 4);    // zero row for conv1
  k_zero_bf<<<1, 256, 0, stream>>>(h1 + (size_t)M * 32, 32);   // zero row for conv2
  k_vox_sum<<<(N * 4 + 255) / 256, 256, 0, stream>>>(pF, idxq, vf, N * 4);
  k_vox_div<<<(M * 4 + 255) / 256, 256, 0, stream>>>(vf, counts, vfb, M * 4);

  int T1 = (M + 15) / 16;
  int B1 = (T1 + 7) / 8; if (B1 > 1024) B1 = 1024; if (B1 < 1) B1 = 1;
  int T2 = (M + 31) / 32;
  int B2 = (T2 + 7) / 8; if (B2 > 1024) B2 = 1024; if (B2 < 1) B2 = 1;
  int Bbn = (int)(((long long)M * 32 + 255) / 256);

  k_conv1<<<B1, 256, 0, stream>>>(vfb, kmap, w1, big, stats + 0, stats + 32, M);
  k_bn_relu<<<Bbn, 256, 0, stream>>>(big, stats + 0, stats + 32, g1, b1, h1, M);
  k_conv2<<<B2, 256, 0, stream>>>(h1, kmap, w2, big, stats + 64, stats + 96, M);
  k_bn_relu<<<Bbn, 256, 0, stream>>>(big, stats + 64, stats + 96, g2, b2, h2, M);

  int TN = (N + 15) / 16;
  int BN_ = (TN + 7) / 8; if (BN_ > 2048) BN_ = 2048; if (BN_ < 1) BN_ = 1;
  k_devox_pt<<<BN_, 256, 0, stream>>>(h2, didx, dwv, w_pt, b_pt, big,
                                      stats + 128, stats + 160, N);
  k_cls<<<BN_, 256, 0, stream>>>(big, stats + 128, stats + 160, g3, b3,
                                 w_cls, b_cls, out, N);
}
